// EchoStateReservoir_52003464020146
// MI455X (gfx1250) — compile-verified, matrix operand order checked by reading, not run on hardware
//
#include <hip/hip_runtime.h>
#include <hip/hip_bf16.h>
#include <math.h>

// ---------------------------------------------------------------------------
// Echo State Reservoir on gfx1250 (MI455X), wave32 + v_wmma_f32_16x16x32_bf16
//
//   1) esn_rho        : power iteration -> scale = rho_target/rho, lk
//   2) esn_scale_cvt / esn_cvt : bf16 operand preparation
//   3) esn_init       : zero h state + global barrier counter
//   4) esn_input_proj : U = x @ W_in^T   (WMMA, 2x2 tiles/wave)
//   5) esn_scan       : persistent 64-WG recurrence, W fragments hoisted
//                       into VGPRs, 4 serial WMMAs/step/wave, LDS reduce,
//                       device-wide barrier, U prefetch for t+1
//   6) esn_readout    : out = H @ readout_W^T (WMMA, 2x2 tiles/wave)
// ---------------------------------------------------------------------------

typedef __bf16 bf16;
typedef __attribute__((ext_vector_type(16))) bf16  v16bf;
typedef __attribute__((ext_vector_type(8)))  bf16  v8bf;
typedef __attribute__((ext_vector_type(8)))  float v8f;

#define DMODEL 512
#define DR     1024
#define BB     8
#define TT     2048

// ---- workspace layout (bytes, 256-aligned) --------------------------------
static const size_t OFF_HDR = 0;                                   // scale, lk, ctr
static const size_t OFF_WB  = 256;                                 // bf16 1024*1024
static const size_t OFF_WI  = OFF_WB + (size_t)DR*DR*2;            // bf16 1024*512
static const size_t OFF_RB  = OFF_WI + (size_t)DR*DMODEL*2;        // bf16 512*1024
static const size_t OFF_XB  = OFF_RB + (size_t)DMODEL*DR*2;        // bf16 8*2048*512
static const size_t OFF_U   = OFF_XB + (size_t)BB*TT*DMODEL*2;     // f32  2048*8*1024
static const size_t OFF_HB  = OFF_U  + (size_t)TT*BB*DR*4;         // bf16 16*1024 (padded h)
static const size_t OFF_HF  = OFF_HB + (size_t)16*DR*2;            // f32  8*1024
static const size_t OFF_HBIG= OFF_HF + (size_t)BB*DR*4;            // bf16 8*2048*1024

// ---- WMMA fragment loader --------------------------------------------------
// Row-major (along K) bf16 matrix. ISA 7.12.2 16-bit A layout (B symmetric):
//   lane L: row = L%16; halves i<8 -> K = k0 + base + i
//                       halves i>=8 -> K = k0 + base + 16 + (i-8)
//   base = 8 for lanes 16..31 else 0.   => two b128 loads per lane.
union V16U { v16bf v; v8bf h[2]; };

__device__ __forceinline__ v16bf ld_frag(const bf16* row, int k0, int hi) {
    const int o = k0 + (hi << 3);
    V16U u;
    u.h[0] = *reinterpret_cast<const v8bf*>(row + o);
    u.h[1] = *reinterpret_cast<const v8bf*>(row + o + 16);
    return u.v;
}

__device__ __forceinline__ v8f wmma_bf16(v16bf a, v16bf b, v8f c) {
    return __builtin_amdgcn_wmma_f32_16x16x32_bf16(
        /*neg_a=*/false, a, /*neg_b=*/false, b,
        /*c_mod=*/(short)0, c, /*reuse_a=*/false, /*reuse_b=*/false);
}

// ---------------------------------------------------------------------------
// 1) power iteration -> hdr[0]=scale, hdr[1]=lk
// ---------------------------------------------------------------------------
__global__ void esn_rho(const float* __restrict__ W, const float* __restrict__ cv,
                        const float* __restrict__ log_rho, const float* __restrict__ leak,
                        float* hdr) {
    __shared__ float sv[DR];
    __shared__ float red[DR];
    const int tid = threadIdx.x;
    sv[tid] = cv[tid];
    __syncthreads();
    const float* row = W + (size_t)tid * DR;
    for (int it = 0; it < 3; ++it) {
        float s = 0.f;
        for (int k = 0; k < DR; ++k) s = fmaf(row[k], sv[k], s);
        red[tid] = s * s;
        __syncthreads();
        for (int off = DR >> 1; off; off >>= 1) {
            if (tid < off) red[tid] += red[tid + off];
            __syncthreads();
        }
        float n = sqrtf(red[0]);
        __syncthreads();
        sv[tid] = (n > 0.f) ? s / n : s;
        __syncthreads();
    }
    float s = 0.f;
    for (int k = 0; k < DR; ++k) s = fmaf(row[k], sv[k], s);
    red[tid] = s * s;
    __syncthreads();
    for (int off = DR >> 1; off; off >>= 1) {
        if (tid < off) red[tid] += red[tid + off];
        __syncthreads();
    }
    if (tid == 0) {
        float rho = sqrtf(red[0]);
        float rho_target = 1.5f / (1.f + expf(-log_rho[0]));
        hdr[0] = rho_target / fmaxf(rho, 1e-6f);
        hdr[1] = 1.f / (1.f + expf(-leak[0]));
    }
}

// ---------------------------------------------------------------------------
// 2) conversions
// ---------------------------------------------------------------------------
__global__ void esn_scale_cvt(const float* __restrict__ src, bf16* __restrict__ dst,
                              const float* __restrict__ hdr, int n) {
    const float sc = hdr[0];
    for (int i = blockIdx.x * blockDim.x + threadIdx.x; i < n; i += gridDim.x * blockDim.x)
        dst[i] = (bf16)(src[i] * sc);
}

__global__ void esn_cvt(const float* __restrict__ src, bf16* __restrict__ dst, int n) {
    for (int i = blockIdx.x * blockDim.x + threadIdx.x; i < n; i += gridDim.x * blockDim.x)
        dst[i] = (bf16)src[i];
}

// ---------------------------------------------------------------------------
// 3) zero init of h state + barrier counter
// ---------------------------------------------------------------------------
__global__ void esn_init(bf16* hB, float* hF, unsigned* ctr) {
    const int i = blockIdx.x * blockDim.x + threadIdx.x;
    for (int j = i; j < 16 * DR; j += gridDim.x * blockDim.x) hB[j] = (bf16)0.f;
    for (int j = i; j < BB * DR; j += gridDim.x * blockDim.x) hF[j] = 0.f;
    if (i == 0) *ctr = 0u;
}

// ---------------------------------------------------------------------------
// 4) U[t*8+b, r] = sum_d x[b,t,d] * W_in[r,d]   (M=16384, N=1024, K=512)
//    2x2 tiles per wave: 8 b128 loads per 4 WMMAs.
//    M-tile mt -> m = mt*16+col -> (t = m>>3, b = m&7); mt+1 shifts t by 2,
//    so the second A row pointer is arow0 + 2*DMODEL (same b).
// ---------------------------------------------------------------------------
__global__ void esn_input_proj(const bf16* __restrict__ Xb, const bf16* __restrict__ Wi,
                               float* __restrict__ U) {
    const int wid  = blockIdx.x * (blockDim.x >> 5) + (threadIdx.x >> 5);
    const int lane = threadIdx.x & 31;
    const int ntp = wid & 31;       // 32 N-tile pairs (64 tiles)
    const int mtp = wid >> 5;       // 512 M-tile pairs (1024 tiles)
    const int col = lane & 15, hi = lane >> 4;

    const int mt0 = mtp * 2, nt0 = ntp * 2;
    const int m = mt0 * 16 + col;           // flattened t*8+b
    const int t = m >> 3, b = m & 7;
    const bf16* arow0 = Xb + ((size_t)b * TT + t) * DMODEL;
    const bf16* arow1 = arow0 + 2 * DMODEL;                      // m+16 -> t+2
    const bf16* brow0 = Wi + (size_t)(nt0 * 16 + col) * DMODEL;
    const bf16* brow1 = brow0 + (size_t)16 * DMODEL;

    v8f acc00 = {}, acc01 = {}, acc10 = {}, acc11 = {};
    #pragma unroll
    for (int k0 = 0; k0 < DMODEL; k0 += 32) {
        v16bf a0 = ld_frag(arow0, k0, hi);
        v16bf a1 = ld_frag(arow1, k0, hi);
        v16bf b0 = ld_frag(brow0, k0, hi);
        v16bf b1 = ld_frag(brow1, k0, hi);
        acc00 = wmma_bf16(a0, b0, acc00);
        acc01 = wmma_bf16(a0, b1, acc01);
        acc10 = wmma_bf16(a1, b0, acc10);
        acc11 = wmma_bf16(a1, b1, acc11);
    }
    #pragma unroll
    for (int j = 0; j < 8; ++j) {
        const int mr0 = mt0 * 16 + j + (hi << 3);
        const int n0  = nt0 * 16 + col;
        U[(size_t)mr0 * DR + n0]            = acc00[j];
        U[(size_t)mr0 * DR + n0 + 16]       = acc01[j];
        U[(size_t)(mr0 + 16) * DR + n0]     = acc10[j];
        U[(size_t)(mr0 + 16) * DR + n0 + 16]= acc11[j];
    }
}

// ---------------------------------------------------------------------------
// 5) persistent scan: 64 WGs (one per 16-wide N-slice) x 8 waves (K split
//    into 128-element chunks). Reservoir weight fragments are LOOP-INVARIANT:
//    hoisted into 32 VGPRs before the time loop, so the per-step critical
//    path is 2 A-loads + 1 WMMA per k-chunk (4 serial WMMAs), LDS reduce,
//    tanh/leaky blend, device-wide barrier.
// ---------------------------------------------------------------------------
__global__ void esn_scan(const bf16* __restrict__ Wb, const float* __restrict__ U,
                         const float* __restrict__ hdr,
                         bf16* hB, float* hF, bf16* __restrict__ Hb, unsigned* ctr) {
    const int bn   = blockIdx.x;         // 0..63 : N tile
    const int w    = threadIdx.x >> 5;   // 0..7  : K chunk
    const int lane = threadIdx.x & 31;
    const int col  = lane & 15, hi = lane >> 4;
    const float lk = hdr[1];

    __shared__ float part[8][32][8];

    const bf16* brow = Wb + (size_t)(bn * 16 + col) * DR;  // weight row n (B[k][n]=W[n][k])
    const bf16* arow = hB + (size_t)col * DR;              // h row m = col (rows 8..15 stay 0)

    // ---- hoist the constant reservoir-weight fragments into VGPRs ----------
    v16bf wfrag[4];
    #pragma unroll
    for (int kk = 0; kk < 4; ++kk)
        wfrag[kk] = ld_frag(brow, w * 128 + kk * 32, hi);

    for (int t = 0; t < TT; ++t) {
        v8f acc = {};
        #pragma unroll
        for (int kk = 0; kk < 4; ++kk) {
            v16bf a = ld_frag(arow, w * 128 + kk * 32, hi);
            acc = wmma_bf16(a, wfrag[kk], acc);
        }
        #pragma unroll
        for (int j = 0; j < 8; ++j) part[w][lane][j] = acc[j];
        __syncthreads();

        if (w == 0 && hi == 0) {            // lanes 0..15 own rows 0..7 (cols = lane)
            const int n = bn * 16 + col;
            #pragma unroll
            for (int j = 0; j < 8; ++j) {   // j = row m = batch b (only m<8 valid)
                float s = 0.f;
                #pragma unroll
                for (int q = 0; q < 8; ++q) s += part[q][lane][j];
                const float u  = U[((size_t)t * BB + j) * DR + n];
                const float hp = hF[j * DR + n];
                const float hn = (1.f - lk) * hp + lk * tanhf(s + u);
                hF[j * DR + n] = hn;
                const bf16 hb = (bf16)hn;
                hB[j * DR + n] = hb;                               // next-step A operand
                Hb[((size_t)j * TT + t) * DR + n] = hb;            // H[b][t][r] for readout
            }
            // prefetch next step's input slice while we sit in the barrier
            if (t + 1 < TT) {
                #pragma unroll
                for (int j = 0; j < 8; ++j)
                    __builtin_prefetch(&U[((size_t)(t + 1) * BB + j) * DR + n], 0, 0);
            }
        }
        __syncthreads();

        // ---- device-wide barrier (monotonic counter, release/acquire) -----
        if (threadIdx.x == 0) {
            __threadfence();                       // release our h writes
            atomicAdd(ctr, 1u);
            const unsigned target = 64u * (unsigned)(t + 1);
            volatile unsigned* vc = ctr;
            while (*vc < target) { __builtin_amdgcn_s_sleep(1); }
        }
        __syncthreads();
        __threadfence();                           // acquire: invalidate stale lines
    }
}

// ---------------------------------------------------------------------------
// 6) out[b,t,d] = sum_r H[b,t,r] * readout_W[d,r]  (M=16384, N=512, K=1024)
//    2x2 tiles per wave.
// ---------------------------------------------------------------------------
__global__ void esn_readout(const bf16* __restrict__ Hb, const bf16* __restrict__ Rb,
                            float* __restrict__ out) {
    const int wid  = blockIdx.x * (blockDim.x >> 5) + (threadIdx.x >> 5);
    const int lane = threadIdx.x & 31;
    const int ntp = wid & 15;       // 16 N-tile pairs (32 tiles)
    const int mtp = wid >> 4;       // 512 M-tile pairs (1024 tiles)
    const int col = lane & 15, hi = lane >> 4;

    const int mt0 = mtp * 2, nt0 = ntp * 2;
    const bf16* arow0 = Hb + (size_t)(mt0 * 16 + col) * DR;   // row m = b*T+t (contiguous)
    const bf16* arow1 = arow0 + (size_t)16 * DR;
    const bf16* brow0 = Rb + (size_t)(nt0 * 16 + col) * DR;   // readout row d
    const bf16* brow1 = brow0 + (size_t)16 * DR;

    v8f acc00 = {}, acc01 = {}, acc10 = {}, acc11 = {};
    #pragma unroll
    for (int k0 = 0; k0 < DR; k0 += 32) {
        v16bf a0 = ld_frag(arow0, k0, hi);
        v16bf a1 = ld_frag(arow1, k0, hi);
        v16bf b0 = ld_frag(brow0, k0, hi);
        v16bf b1 = ld_frag(brow1, k0, hi);
        acc00 = wmma_bf16(a0, b0, acc00);
        acc01 = wmma_bf16(a0, b1, acc01);
        acc10 = wmma_bf16(a1, b0, acc10);
        acc11 = wmma_bf16(a1, b1, acc11);
    }
    #pragma unroll
    for (int j = 0; j < 8; ++j) {
        const int mr0 = mt0 * 16 + j + (hi << 3);
        const int n0  = nt0 * 16 + col;
        out[(size_t)mr0 * DMODEL + n0]             = acc00[j];
        out[(size_t)mr0 * DMODEL + n0 + 16]        = acc01[j];
        out[(size_t)(mr0 + 16) * DMODEL + n0]      = acc10[j];
        out[(size_t)(mr0 + 16) * DMODEL + n0 + 16] = acc11[j];
    }
}

// ---------------------------------------------------------------------------
extern "C" void kernel_launch(void* const* d_in, const int* in_sizes, int n_in,
                              void* d_out, int out_size, void* d_ws, size_t ws_size,
                              hipStream_t stream) {
    const float* x        = (const float*)d_in[0];   // [8,2048,512]
    const float* W_in     = (const float*)d_in[1];   // [1024,512]
    const float* W_res    = (const float*)d_in[2];   // [1024,1024]
    const float* read_W   = (const float*)d_in[3];   // [512,1024]
    const float* log_rho  = (const float*)d_in[4];   // scalar
    const float* leak     = (const float*)d_in[5];   // scalar
    const float* cached_v = (const float*)d_in[6];   // [1024]
    float* out = (float*)d_out;

    char* ws = (char*)d_ws;
    float*    hdr = (float*)   (ws + OFF_HDR);
    unsigned* ctr = (unsigned*)(ws + OFF_HDR + 8);
    bf16*  Wb = (bf16*)(ws + OFF_WB);
    bf16*  Wi = (bf16*)(ws + OFF_WI);
    bf16*  Rb = (bf16*)(ws + OFF_RB);
    bf16*  Xb = (bf16*)(ws + OFF_XB);
    float* U  = (float*)(ws + OFF_U);
    bf16*  hB = (bf16*)(ws + OFF_HB);
    float* hF = (float*)(ws + OFF_HF);
    bf16*  Hb = (bf16*)(ws + OFF_HBIG);

    // 1) spectral-radius scale + leak
    esn_rho<<<1, DR, 0, stream>>>(W_res, cached_v, log_rho, leak, hdr);

    // 2) precision conversion (bf16 operands for WMMA)
    esn_scale_cvt<<<1024, 256, 0, stream>>>(W_res, Wb, hdr, DR * DR);
    esn_cvt<<<512, 256, 0, stream>>>(W_in, Wi, DR * DMODEL);
    esn_cvt<<<512, 256, 0, stream>>>(read_W, Rb, DMODEL * DR);
    esn_cvt<<<2048, 256, 0, stream>>>(x, Xb, BB * TT * DMODEL);

    // 3) zero h state + barrier counter
    esn_init<<<32, 256, 0, stream>>>(hB, hF, ctr);

    // 4) input projection GEMM: 16384 waves (2x2 tiles each) / 8 per block
    esn_input_proj<<<2048, 256, 0, stream>>>(Xb, Wi, U);

    // 5) persistent sequential scan: 64 WGs, 2048 steps
    esn_scan<<<64, 256, 0, stream>>>(Wb, U, hdr, hB, hF, Hb, ctr);

    // 6) readout GEMM: 8192 waves (2x2 tiles each) / 8 per block
    esn_readout<<<1024, 256, 0, stream>>>(Hb, Rb, out);
}
